// I2HOFI_49194555408496
// MI455X (gfx1250) — matrix operand order repaired and checked by reading, not compiled, not run on hardware
//
#include <hip/hip_runtime.h>

// ---------------- problem constants ----------------
#define Bsz   16
#define Rr    13
#define Nn    392
#define Ff    256
#define PCH   512
#define NCLS  200
#define ALPHA 0.5f
#define BN_EPS 1e-3f

typedef __attribute__((ext_vector_type(16))) __bf16 v16bf;
typedef __attribute__((ext_vector_type(8)))  __bf16 v8bf;
typedef __attribute__((ext_vector_type(4)))  __bf16 v4bf;
typedef __attribute__((ext_vector_type(8)))  float  v8f;

static __device__ __forceinline__ __bf16 f2bf(float f) { return (__bf16)f; }

// two contiguous 16B runs -> one 16-element fragment (2x b128 loads)
static __device__ __forceinline__ v16bf ld_frag2(const __bf16* lo, const __bf16* hi) {
  v8bf a = *(const v8bf*)lo;
  v8bf b = *(const v8bf*)hi;
  return __builtin_shufflevector(a, b, 0, 1, 2, 3, 4, 5, 6, 7,
                                 8, 9, 10, 11, 12, 13, 14, 15);
}
// one contiguous 32B run -> 16-element fragment
static __device__ __forceinline__ v16bf ld_frag16(const __bf16* p) {
  return ld_frag2(p, p + 8);
}

static __device__ __forceinline__ v8f wmma_bf16(v16bf a, v16bf b, v8f c) {
  // D = A(16x32 bf16) * B(32x16 bf16) + C(16x16 f32)
  return __builtin_amdgcn_wmma_f32_16x16x32_bf16(false, a, false, b, (short)0, c,
                                                 false, false);
}

static __device__ __forceinline__ float lrelu(float x) { return x > 0.f ? x : 0.2f * x; }
static __device__ __forceinline__ float sigm(float x)  { return 1.f / (1.f + __expf(-x)); }

// A-fragment K index for element e (lane half kb in {0,8}):
// e=0..7 -> kb+e ; e=8..15 -> kb+16+(e-8)   (two contiguous runs)
static __device__ __forceinline__ int a_kidx(int e, int kb) {
  return kb + ((e < 8) ? e : (16 + e - 8));
}

// ============================================================
// Generic GEMM: C[M,N] = act(A[M,K] @ W[K,N] + bias), bf16 WMMA
// REQUIRES: M % 64 == 0, N % 64 == 0, K % 32 == 0 (true for all uses)
// block = 128 threads (4 waves), tile 64x64, wave w -> rows 16w..16w+15
// LDS: A tile row-major [row][K32]; W tile transposed [col][K32]
// ============================================================
__global__ __launch_bounds__(128)
void gemm_act(const float* __restrict__ A, const float* __restrict__ W,
              const float* __restrict__ bias, float* __restrict__ C,
              int M, int K, int N, int act) {
  __shared__ __align__(16) __bf16 As[64][32];
  __shared__ __align__(16) __bf16 Wt[64][32];      // [col][K]
  const int t = threadIdx.x, lane = t & 31, wv = t >> 5;
  const int tm = blockIdx.x * 64;
  const int tn = blockIdx.y * 64;

  v8f acc[4];
#pragma unroll
  for (int nt = 0; nt < 4; ++nt)
#pragma unroll
    for (int v = 0; v < 8; ++v) acc[nt][v] = 0.f;

  const int kb  = (lane < 16) ? 0 : 8;
  const int kb2 = (lane < 16) ? 0 : 16;

  for (int k0 = 0; k0 < K; k0 += 32) {
    __syncthreads();
#pragma unroll
    for (int i = 0; i < 4; ++i) {                  // A tile 64x32: float4 loads
      int idx = i * 128 + t;                       // 0..511 float4s
      int r = idx >> 3, c4 = (idx & 7) * 4;
      float4 p = *(const float4*)(A + (size_t)(tm + r) * K + k0 + c4);
      v4bf q = { f2bf(p.x), f2bf(p.y), f2bf(p.z), f2bf(p.w) };
      *(v4bf*)&As[r][c4] = q;                      // 8B contiguous store
    }
#pragma unroll
    for (int i = 0; i < 4; ++i) {                  // W tile 32x64 -> transposed
      int idx = i * 128 + t;                       // 0..511
      int c = idx & 63, r0 = (idx >> 6) * 4;
      const float* wp = W + (size_t)(k0 + r0) * N + tn + c;
      v4bf q = { f2bf(wp[0]), f2bf(wp[(size_t)N]),
                 f2bf(wp[(size_t)2 * N]), f2bf(wp[(size_t)3 * N]) };
      *(v4bf*)&Wt[c][r0] = q;                      // 8B contiguous store
    }
    __syncthreads();

    const int arow = wv * 16 + (lane & 15);
    v16bf a = ld_frag16(&As[arow][kb2]);
#pragma unroll
    for (int nt = 0; nt < 4; ++nt) {
      v16bf bb = ld_frag16(&Wt[nt * 16 + (lane & 15)][kb2]);
      acc[nt] = wmma_bf16(a, bb, acc[nt]);
    }
  }

#pragma unroll
  for (int nt = 0; nt < 4; ++nt)
#pragma unroll
    for (int v = 0; v < 8; ++v) {
      int row = tm + wv * 16 + v + ((lane < 16) ? 0 : 8);
      int col = tn + nt * 16 + (lane & 15);
      float x = acc[nt][v] + (bias ? bias[col] : 0.f);
      if (act == 1) x = sigm(x);
      C[(size_t)row * N + col] = x;
    }
}

// ============================================================
// APPNP with rank-1 adjacency: out = (1-a)(S+h)/(n+1) + a*h + x
// ============================================================
__global__ void appnp_intra(const float* __restrict__ H, const float* __restrict__ X,
                            float* __restrict__ O) {
  int g = blockIdx.x;               // b*R + r
  int f = threadIdx.x;              // 0..255
  size_t base = (size_t)g * Nn * Ff + f;
  float S = 0.f;
  for (int n = 0; n < Nn; ++n) S += H[base + (size_t)n * Ff];
  const float c1 = (1.f - ALPHA) / (Nn + 1.f);
  for (int n = 0; n < Nn; ++n) {
    size_t idx = base + (size_t)n * Ff;
    float h = H[idx];
    O[idx] = c1 * (S + h) + ALPHA * h + X[idx];
  }
}

__global__ void appnp_inter(const float* __restrict__ H, const float* __restrict__ X,
                            float* __restrict__ O) {
  int bn = blockIdx.x;              // b*N + n
  int b = bn / Nn, n = bn % Nn;
  int f = threadIdx.x;
  float S = 0.f;
  for (int r = 0; r < Rr; ++r) S += H[(((size_t)(b * Rr + r)) * Nn + n) * Ff + f];
  const float c2 = (1.f - ALPHA) / (Rr + 1.f);
  for (int r = 0; r < Rr; ++r) {
    size_t src = (((size_t)(b * Rr + r)) * Nn + n) * Ff + f;
    size_t dst = (((size_t)(b * Nn + n)) * Rr + r) * Ff + f;
    float h = H[src];
    O[dst] = c2 * (S + h) + ALPHA * h + X[src];
  }
}

// ============================================================
// s[row] = xk . a_self ; nb[row] = xk . a_neigh   (wave per row)
// ============================================================
__global__ void attn_scores(const float* __restrict__ XK, const float* __restrict__ aS,
                            const float* __restrict__ aN, float* __restrict__ s,
                            float* __restrict__ nb, int M) {
  int lane = threadIdx.x & 31, wv = threadIdx.x >> 5;
  int row = blockIdx.x * 8 + wv;
  if (row >= M) return;
  const float* x = XK + (size_t)row * Ff;
  float ds = 0.f, dn = 0.f;
  for (int f = lane; f < Ff; f += 32) { float v = x[f]; ds += v * aS[f]; dn += v * aN[f]; }
#pragma unroll
  for (int off = 16; off; off >>= 1) {
    ds += __shfl_xor(ds, off, 32);
    dn += __shfl_xor(dn, off, 32);
  }
  if (lane == 0) { s[row] = ds; nb[row] = dn; }
}

__global__ void max_nb_graph(const float* __restrict__ nb, float* __restrict__ mx,
                             int nodes) {
  __shared__ float red[256];
  int g = blockIdx.x, t = threadIdx.x;
  float m = -1e30f;
  for (int i = t; i < nodes; i += 256) m = fmaxf(m, nb[(size_t)g * nodes + i]);
  red[t] = m;
  __syncthreads();
  for (int s2 = 128; s2; s2 >>= 1) {
    if (t < s2) red[t] = fmaxf(red[t], red[t + s2]);
    __syncthreads();
  }
  if (t == 0) mx[g] = red[0];
}

// ============================================================
// Intra-ROI GAT: graphs (b,r), 392 nodes. One-pass softmax attention
// using monotone leaky_relu rowmax trick; P tiles built as WMMA A-frags.
// xk tile staged transposed in LDS: XsT[col][m32] -> contiguous fragments.
// Output x2 stored bf16 (feeds the bf16 pooling GEMM directly).
// block = 128 thr (4 waves = 64 output rows), grid = (208, 7)
// ============================================================
__global__ __launch_bounds__(128)
void gat_intra(const float* __restrict__ XK, const float* __restrict__ s,
               const float* __restrict__ nb, const float* __restrict__ mxnb,
               const float* __restrict__ bias, const float* __restrict__ Hres,
               __bf16* __restrict__ x2) {
  __shared__ __align__(16) __bf16 XsT[Ff][32];     // 16 KB, [col][m]
  const int g = blockIdx.x;                        // b*R + r
  const int b = g / Rr, r = g % Rr;
  const int t = threadIdx.x, lane = t & 31, wv = t >> 5;
  const int n0 = blockIdx.y * 64 + wv * 16;
  const size_t grow = (size_t)g * Nn;
  const int nrow = n0 + (lane & 15);
  const bool nok = nrow < Nn;
  const int nclamp = nok ? nrow : (Nn - 1);

  const float sval   = s[grow + nclamp];
  const float rowmax = lrelu(sval + mxnb[g]);
  const int kb  = (lane < 16) ? 0 : 8;
  const int kb2 = (lane < 16) ? 0 : 16;

  v8f acc[16];
#pragma unroll
  for (int ft = 0; ft < 16; ++ft)
#pragma unroll
    for (int v = 0; v < 8; ++v) acc[ft][v] = 0.f;
  float dsum = 0.f;

  for (int m0 = 0; m0 < Nn; m0 += 32) {
    __syncthreads();
#pragma unroll
    for (int i = 0; i < 16; ++i) {                 // stage xk[m0..m0+31,:] transposed
      int idx = i * 128 + t;                       // 0..2047 quads
      int c = idx & 255, rq = (idx >> 8) * 4;      // rq in {0,4,...,28}
      v4bf q;
#pragma unroll
      for (int j = 0; j < 4; ++j) {
        int m = m0 + rq + j;
        int mc = (m < Nn) ? m : (Nn - 1);          // clamp + select: no branches
        float v = XK[(grow + mc) * Ff + c];
        q[j] = f2bf((m < Nn) ? v : 0.f);
      }
      *(v4bf*)&XsT[c][rq] = q;                     // 8B contiguous store
    }
    __syncthreads();

    v16bf a;                                       // P tile as A-fragment
#pragma unroll
    for (int e = 0; e < 16; ++e) {
      int m = m0 + kb2 + e;
      int mc = (m < Nn) ? m : (Nn - 1);
      float nbv = nb[grow + mc];
      float p = (m < Nn && nok) ? __expf(lrelu(sval + nbv) - rowmax) : 0.f;
      dsum += p;
      a[e] = f2bf(p);
    }
#pragma unroll
    for (int ft = 0; ft < 16; ++ft) {
      v16bf bb = ld_frag16(&XsT[ft * 16 + (lane & 15)][kb2]);
      acc[ft] = wmma_bf16(a, bb, acc[ft]);
    }
  }

  dsum += __shfl_xor(dsum, 16, 32);                // full row denominator

#pragma unroll
  for (int ft = 0; ft < 16; ++ft)
#pragma unroll
    for (int v = 0; v < 8; ++v) {
      int Mr = v + ((lane < 16) ? 0 : 8);
      int n = n0 + Mr;
      if (n < Nn) {
        int col = ft * 16 + (lane & 15);
        float d = __shfl(dsum, Mr, 32);            // lane Mr holds row Mr's denom
        float val = acc[ft][v] / d + bias[col];
        val = (val > 0.f) ? val : (__expf(val) - 1.f);   // elu
        size_t ridx = (grow + n) * Ff + col;
        val += Hres[ridx];                         // GAT residual
        size_t orow = (size_t)b * (2 * Rr * Nn) + (size_t)r * Nn + n;
        x2[orow * Ff + col] = f2bf(val);
      }
    }
}

// ============================================================
// Inter-ROI GAT: graphs (b,n), 13 nodes -> one wave per graph,
// single 32-wide K chunk (padded). block = 4 waves = 4 graphs.
// ============================================================
__global__ __launch_bounds__(128)
void gat_inter(const float* __restrict__ XK, const float* __restrict__ s,
               const float* __restrict__ nb, const float* __restrict__ mxnb,
               const float* __restrict__ bias, const float* __restrict__ Hres,
               __bf16* __restrict__ x2) {
  const int t = threadIdx.x, lane = t & 31, wv = t >> 5;
  const int g = blockIdx.x * 4 + wv;               // b*N + n  (grid exact)
  const int b = g / Nn, n = g % Nn;
  const size_t grow = (size_t)g * Rr;
  const int rrow = lane & 15;
  const bool rok = rrow < Rr;
  const int rclamp = rok ? rrow : (Rr - 1);

  const float sval   = s[grow + rclamp];
  const float rowmax = lrelu(sval + mxnb[g]);
  const int kb  = (lane < 16) ? 0 : 8;
  const int kb2 = (lane < 16) ? 0 : 16;

  v16bf a;
  float dsum = 0.f;
#pragma unroll
  for (int e = 0; e < 16; ++e) {
    int m = kb2 + e;
    int mc = (m < Rr) ? m : (Rr - 1);
    float nbv = nb[grow + mc];
    float p = (m < Rr && rok) ? __expf(lrelu(sval + nbv) - rowmax) : 0.f;
    dsum += p;
    a[e] = f2bf(p);
  }
  dsum += __shfl_xor(dsum, 16, 32);

  v8f acc[16];
#pragma unroll
  for (int ft = 0; ft < 16; ++ft) {
#pragma unroll
    for (int v = 0; v < 8; ++v) acc[ft][v] = 0.f;
    v16bf bb;
    int col = ft * 16 + (lane & 15);
#pragma unroll
    for (int e = 0; e < 16; ++e) {
      int m = kb2 + e;
      int mc = (m < Rr) ? m : (Rr - 1);
      float v = XK[(grow + mc) * Ff + col];
      bb[e] = f2bf((m < Rr) ? v : 0.f);
    }
    acc[ft] = wmma_bf16(a, bb, acc[ft]);
  }

#pragma unroll
  for (int ft = 0; ft < 16; ++ft)
#pragma unroll
    for (int v = 0; v < 8; ++v) {
      int Mr = v + ((lane < 16) ? 0 : 8);
      if (Mr < Rr) {
        int col = ft * 16 + (lane & 15);
        float d = __shfl(dsum, Mr, 32);
        float val = acc[ft][v] / d + bias[col];
        val = (val > 0.f) ? val : (__expf(val) - 1.f);
        size_t ridx = (grow + Mr) * Ff + col;
        val += Hres[ridx];
        size_t orow = (size_t)b * (2 * Rr * Nn) + (size_t)(Rr * Nn) + (size_t)n * Rr + Mr;
        x2[orow * Ff + col] = f2bf(val);
      }
    }
}

// ============================================================
// Fused GlobalAttentionPool: pooled[b,c] = sum_rows sigmoid(x2@wa+ba)*(x2@wf+bf)
// x2 already bf16 -> A-fragments are two global b128 loads, zero conversion.
// Weights staged once, transposed [col][K256] in LDS.
// ROWS is a multiple of 16 -> row guards are wave-uniform scalar branches.
// block = 128 thr, 32 output cols per block.
// ============================================================
__global__ __launch_bounds__(128)
void pool_fused(const __bf16* __restrict__ x2, const float* __restrict__ wf,
                const float* __restrict__ bfp, const float* __restrict__ wa,
                const float* __restrict__ bap, float* __restrict__ pooled) {
  __shared__ __align__(16) __bf16 WaT[32][Ff];     // [col][K]
  __shared__ __align__(16) __bf16 WfT[32][Ff];
  __shared__ float red[2][4][16];
  const int t = threadIdx.x, lane = t & 31, wv = t >> 5;
  const int b = blockIdx.y;
  const int c0 = blockIdx.x * 32;
  const int ROWS = 2 * Rr * Nn;                    // 10192 (multiple of 16)

#pragma unroll
  for (int i = 0; i < 16; ++i) {                   // 2048 K-quads per matrix
    int idx = i * 128 + t;
    int c = idx & 31, r0 = (idx >> 5) * 4;
    const float* ap = wa + (size_t)r0 * PCH + c0 + c;
    const float* fp = wf + (size_t)r0 * PCH + c0 + c;
    v4bf qa = { f2bf(ap[0]), f2bf(ap[PCH]), f2bf(ap[2 * PCH]), f2bf(ap[3 * PCH]) };
    v4bf qf = { f2bf(fp[0]), f2bf(fp[PCH]), f2bf(fp[2 * PCH]), f2bf(fp[3 * PCH]) };
    *(v4bf*)&WaT[c][r0] = qa;
    *(v4bf*)&WfT[c][r0] = qf;
  }
  __syncthreads();

  float prod[2][8];
#pragma unroll
  for (int nt = 0; nt < 2; ++nt)
#pragma unroll
    for (int v = 0; v < 8; ++v) prod[nt][v] = 0.f;

  const int kb  = (lane < 16) ? 0 : 8;
  const int kb2 = (lane < 16) ? 0 : 16;

  for (int rt = 0; rt < ROWS; rt += 64) {
    int m0 = rt + wv * 16;                         // wave-uniform
    if (m0 < ROWS) {                               // uniform scalar branch
      int row = m0 + (lane & 15);
      v8f accA[2], accF[2];
#pragma unroll
      for (int nt = 0; nt < 2; ++nt)
#pragma unroll
        for (int v = 0; v < 8; ++v) { accA[nt][v] = 0.f; accF[nt][v] = 0.f; }

      const __bf16* xr = x2 + ((size_t)b * ROWS + row) * Ff;
      for (int k0 = 0; k0 < Ff; k0 += 32) {
        // A-fragment: one contiguous 32B run straight from global bf16
        v16bf a = ld_frag16(xr + k0 + kb2);
#pragma unroll
        for (int nt = 0; nt < 2; ++nt) {
          int col = nt * 16 + (lane & 15);
          v16bf bwa = ld_frag16(&WaT[col][k0 + kb2]);
          v16bf bwf = ld_frag16(&WfT[col][k0 + kb2]);
          accA[nt] = wmma_bf16(a, bwa, accA[nt]);
          accF[nt] = wmma_bf16(a, bwf, accF[nt]);
        }
      }
#pragma unroll
      for (int nt = 0; nt < 2; ++nt)
#pragma unroll
        for (int v = 0; v < 8; ++v) {
          int col = c0 + nt * 16 + (lane & 15);
          float at = sigm(accA[nt][v] + bap[col]);
          float fe = accF[nt][v] + bfp[col];
          prod[nt][v] += at * fe;
        }
    }
  }

#pragma unroll
  for (int nt = 0; nt < 2; ++nt) {
    float sme = 0.f;
#pragma unroll
    for (int v = 0; v < 8; ++v) sme += prod[nt][v];
    sme += __shfl_xor(sme, 16, 32);
    if (lane < 16) red[nt][wv][lane] = sme;
  }
  __syncthreads();
  if (t < 32) {
    int nt = t >> 4, cl = t & 15;
    float tot = red[nt][0][cl] + red[nt][1][cl] + red[nt][2][cl] + red[nt][3][cl];
    pooled[(size_t)b * PCH + c0 + nt * 16 + cl] = tot;
  }
}

// ============================================================
// BatchNorm (inference) + classifier + softmax. One block per batch.
// ============================================================
__global__ void head(const float* __restrict__ pooled, const float* __restrict__ gamma,
                     const float* __restrict__ beta, const float* __restrict__ mean,
                     const float* __restrict__ var, const float* __restrict__ Wo,
                     const float* __restrict__ bo, float* __restrict__ out) {
  __shared__ float bn[PCH];
  __shared__ float red[256];
  const int b = blockIdx.x, t = threadIdx.x;
  for (int i = t; i < PCH; i += 256) {
    float p = pooled[(size_t)b * PCH + i];
    bn[i] = (p - mean[i]) * rsqrtf(var[i] + BN_EPS) * gamma[i] + beta[i];
  }
  __syncthreads();
  float lg = -1e30f;
  if (t < NCLS) {
    float acc = bo[t];
    for (int k = 0; k < PCH; ++k) acc += bn[k] * Wo[(size_t)k * NCLS + t];
    lg = acc;
  }
  red[t] = lg;
  __syncthreads();
  for (int s2 = 128; s2; s2 >>= 1) {
    if (t < s2) red[t] = fmaxf(red[t], red[t + s2]);
    __syncthreads();
  }
  float mx = red[0];
  __syncthreads();
  float ex = (t < NCLS) ? __expf(lg - mx) : 0.f;
  red[t] = ex;
  __syncthreads();
  for (int s2 = 128; s2; s2 >>= 1) {
    if (t < s2) red[t] += red[t + s2];
    __syncthreads();
  }
  float sum = red[0];
  if (t < NCLS) out[(size_t)b * NCLS + t] = ex / sum;
}

// ============================================================
extern "C" void kernel_launch(void* const* d_in, const int* in_sizes, int n_in,
                              void* d_out, int out_size, void* d_ws, size_t ws_size,
                              hipStream_t stream) {
  const float* x       = (const float*)d_in[0];
  const float* w_mlp   = (const float*)d_in[1];
  const float* b_mlp   = (const float*)d_in[2];
  const float* k_gat   = (const float*)d_in[3];   // (F,1,F) == [256,256]
  const float* a_self  = (const float*)d_in[4];   // [256]
  const float* a_neigh = (const float*)d_in[5];   // [256]
  const float* b_gat   = (const float*)d_in[6];
  const float* wf_pool = (const float*)d_in[7];
  const float* bf_pool = (const float*)d_in[8];
  const float* wa_pool = (const float*)d_in[9];
  const float* ba_pool = (const float*)d_in[10];
  const float* bn_g    = (const float*)d_in[11];
  const float* bn_b    = (const float*)d_in[12];
  const float* bn_m    = (const float*)d_in[13];
  const float* bn_v    = (const float*)d_in[14];
  const float* w_out   = (const float*)d_in[15];
  const float* b_out   = (const float*)d_in[16];

  float* ws = (float*)d_ws;
  const size_t S1 = (size_t)Bsz * Rr * Nn * Ff;   // 20,873,216 floats
  const int Mrows = Bsz * Rr * Nn;                // 81536 (== Bsz*Nn*Rr)

  float*  H       = ws;               // [B,R,N,F] sigmoid MLP output
  float*  Hintra  = ws + 1 * S1;      // APPNP intra + residual
  float*  Hinter  = ws + 2 * S1;      // APPNP inter + residual ([B,N,R,F])
  float*  XKintra = ws;               // reuse H (dead after APPNP kernels)
  float*  XKinter = ws + 3 * S1;
  __bf16* x2      = (__bf16*)(ws + 4 * S1);  // [B, 2RN, F] bf16 (occupies S1 floats)
  float*  sm      = ws + 5 * S1;
  float*  s_in    = sm;               // 81536
  float*  nb_in   = sm + (size_t)Mrows;
  float*  s_ie    = sm + (size_t)2 * Mrows;
  float*  nb_ie   = sm + (size_t)3 * Mrows;
  float*  mx_in   = sm + (size_t)4 * Mrows;       // 208
  float*  mx_ie   = mx_in + 256;                  // 6272
  float*  pooled  = mx_ie + 8192;                 // 16*512

  dim3 blk(128);

  // 1) H = sigmoid(X @ w_mlp + b_mlp)
  gemm_act<<<dim3(Mrows / 64, Ff / 64), blk, 0, stream>>>(x, w_mlp, b_mlp, H,
                                                          Mrows, Ff, Ff, 1);
  // 2) APPNP (rank-1 adjacency) + residual, both orientations
  appnp_intra<<<Bsz * Rr, 256, 0, stream>>>(H, x, Hintra);
  appnp_inter<<<Bsz * Nn, 256, 0, stream>>>(H, x, Hinter);

  // 3) xk = h @ k_gat
  gemm_act<<<dim3(Mrows / 64, Ff / 64), blk, 0, stream>>>(Hintra, k_gat, nullptr,
                                                          XKintra, Mrows, Ff, Ff, 0);
  gemm_act<<<dim3(Mrows / 64, Ff / 64), blk, 0, stream>>>(Hinter, k_gat, nullptr,
                                                          XKinter, Mrows, Ff, Ff, 0);

  // 4) attention scalars s, nb and per-graph max(nb)
  attn_scores<<<Mrows / 8, 256, 0, stream>>>(XKintra, a_self, a_neigh, s_in, nb_in, Mrows);
  attn_scores<<<Mrows / 8, 256, 0, stream>>>(XKinter, a_self, a_neigh, s_ie, nb_ie, Mrows);
  max_nb_graph<<<Bsz * Rr, 256, 0, stream>>>(nb_in, mx_in, Nn);
  max_nb_graph<<<Bsz * Nn, 256, 0, stream>>>(nb_ie, mx_ie, Rr);

  // 5) GAT attention (single-pass softmax, WMMA P@xk) + bias + elu + residual
  gat_intra<<<dim3(Bsz * Rr, (Nn + 63) / 64), blk, 0, stream>>>(
      XKintra, s_in, nb_in, mx_in, b_gat, Hintra, x2);
  gat_inter<<<(Bsz * Nn) / 4, blk, 0, stream>>>(
      XKinter, s_ie, nb_ie, mx_ie, b_gat, Hinter, x2);

  // 6) fused GlobalAttentionPool (x2 consumed as bf16)
  pool_fused<<<dim3(PCH / 32, Bsz), blk, 0, stream>>>(x2, wf_pool, bf_pool,
                                                      wa_pool, ba_pool, pooled);
  // 7) BN + classifier + softmax
  head<<<Bsz, 256, 0, stream>>>(pooled, bn_g, bn_b, bn_m, bn_v, w_out, b_out,
                                (float*)d_out);
}
